// MambaLayer_27779848470954
// MI455X (gfx1250) — matrix operand order repaired and checked by reading, not compiled, not run on hardware
//
#include <hip/hip_runtime.h>
#include <hip/hip_bf16.h>
#include <math.h>

#define BSZ     4
#define DMODEL  256
#define LSEQ    4096
#define DINNER  512
#define DSTATE  16
#define MROWS   (BSZ * LSEQ)   /* 16384 */

typedef __attribute__((ext_vector_type(16))) __bf16 v16bf;
typedef __attribute__((ext_vector_type(8)))  __bf16 v8bf;
typedef __attribute__((ext_vector_type(8)))  float  v8f;

// ---------------------------------------------------------------------------
// Core WMMA tile accumulator: D(16x16,f32) += A(16xK,bf16) * W(NxK,bf16)^T
// A row-major (lda), W row-major (ldw == per-row K stride). K % 32 == 0.
// A operand k order used here: lane m=lane&15, halves A[m, k+16*(lane>>4)+j] j=0..15
// (contiguous, matching B).
// B operand k order used here: lane n=lane&15, halves B[k+16*(lane>>4)+j, n] j=0..15,
// which for W(N,K) row-major is 16 contiguous halves of row n.
// ---------------------------------------------------------------------------
__device__ __forceinline__ v8f wmma_acc_tile(const __bf16* __restrict__ A, int lda,
                                             const __bf16* __restrict__ W, int ldw,
                                             int m0, int n0, int K) {
  const int lane = threadIdx.x & 31;
  const int r16  = lane & 15;
  const int hi   = lane >> 4;
  const __bf16* pa = A + (size_t)(m0 + r16) * lda + hi * 16;
  const __bf16* pb = W + (size_t)(n0 + r16) * ldw + hi * 16;
  v8f acc = {};
  for (int k = 0; k < K; k += 32) {
    v16bf a, b;
    ((v8bf*)&a)[0] = *(const v8bf*)(pa);
    ((v8bf*)&a)[1] = *(const v8bf*)(pa + 8);
    ((v8bf*)&b)[0] = *(const v8bf*)(pb);
    ((v8bf*)&b)[1] = *(const v8bf*)(pb + 8);
    acc = __builtin_amdgcn_wmma_f32_16x16x32_bf16(false, a, false, b,
                                                  (short)0, acc, false, false);
    pa += 32; pb += 32;
  }
  return acc;
}

// Generic GEMM: C(M,N) f32 = A(M,K)bf16 @ W(N,K)bf16^T.  128 thr = 4 waves,
// wave -> one 16x16 tile; grid.x = N/16, grid.y = M/64.
__global__ void gemm_bf16_f32(const __bf16* __restrict__ A, int lda,
                              const __bf16* __restrict__ W, int ldw,
                              float* __restrict__ C, int N, int K) {
  const int wave = threadIdx.x >> 5;
  const int m0 = (blockIdx.y * 4 + wave) * 16;
  const int n0 = blockIdx.x * 16;
  v8f acc = wmma_acc_tile(A, lda, W, ldw, m0, n0, K);
  const int lane = threadIdx.x & 31;
  const int col  = n0 + (lane & 15);
  const int rb   = m0 + (lane >> 4) * 8;
#pragma unroll
  for (int v = 0; v < 8; ++v)
    C[(size_t)(rb + v) * N + col] = acc[v];
}

// out_proj GEMM fused with skip: xm = y@Wout^T + skip * xf, emitted as bf16
// for the following GEMM. xf[b,l,n] read directly from NCHW input x.
__global__ void gemm_outproj(const __bf16* __restrict__ A,
                             const __bf16* __restrict__ W,
                             const float* __restrict__ x,
                             const float* __restrict__ skip_scale,
                             __bf16* __restrict__ xm_bf) {
  const int wave = threadIdx.x >> 5;
  const int m0 = (blockIdx.y * 4 + wave) * 16;
  const int n0 = blockIdx.x * 16;
  v8f acc = wmma_acc_tile(A, DINNER, W, DINNER, m0, n0, DINNER);
  const float sk = skip_scale[0];
  const int lane = threadIdx.x & 31;
  const int col  = n0 + (lane & 15);
  const int rb   = m0 + (lane >> 4) * 8;
#pragma unroll
  for (int v = 0; v < 8; ++v) {
    const int m = rb + v;
    const int b = m >> 12, l = m & (LSEQ - 1);
    const float sv = sk * x[(((size_t)b * DMODEL + col) << 12) + l];
    xm_bf[(size_t)m * DMODEL + col] = (__bf16)(acc[v] + sv);
  }
}

// Final proj GEMM fused with bias and NHWC->NCHW transposed store to d_out.
__global__ void gemm_proj(const __bf16* __restrict__ A,
                          const __bf16* __restrict__ W,
                          const float* __restrict__ bias,
                          float* __restrict__ out) {
  const int wave = threadIdx.x >> 5;
  const int m0 = (blockIdx.y * 4 + wave) * 16;
  const int n0 = blockIdx.x * 16;
  v8f acc = wmma_acc_tile(A, DMODEL, W, DMODEL, m0, n0, DMODEL);
  const int lane = threadIdx.x & 31;
  const int col  = n0 + (lane & 15);
  const int rb   = m0 + (lane >> 4) * 8;
  const float bv = bias[col];
#pragma unroll
  for (int v = 0; v < 8; ++v) {
    const int m = rb + v;
    const int b = m >> 12, l = m & (LSEQ - 1);
    out[(((size_t)b * DMODEL + col) << 12) + l] = acc[v] + bv;
  }
}

// LayerNorm over channels with LDS transpose (x is NCHW -> channel stride 16KB).
// Block: 256 thr handles one b and 32 consecutive l.
__global__ void layernorm_to_bf16(const float* __restrict__ x,
                                  const float* __restrict__ nw,
                                  const float* __restrict__ nb,
                                  __bf16* __restrict__ xn) {
  __shared__ float sh[32 * 257];
  const int b   = blockIdx.y;
  const int l0  = blockIdx.x * 32;
  const int tid = threadIdx.x;
  const float* xb = x + (size_t)b * DMODEL * LSEQ;
#pragma unroll 4
  for (int it = 0; it < 32; ++it) {
    const int e = it * 256 + tid;
    const int c = e >> 5, li = e & 31;             // coalesced along l
    sh[li * 257 + c] = xb[(size_t)c * LSEQ + l0 + li];
  }
  __syncthreads();
  const int r = tid >> 3, sub = tid & 7, cb = sub * 32;
  float s = 0.f, s2 = 0.f;
#pragma unroll
  for (int j = 0; j < 32; ++j) {
    const float v = sh[r * 257 + cb + j];
    s += v; s2 += v * v;
  }
#pragma unroll
  for (int msk = 1; msk < 8; msk <<= 1) {
    s  += __shfl_xor(s,  msk, 8);
    s2 += __shfl_xor(s2, msk, 8);
  }
  const float mu  = s * (1.f / 256.f);
  const float var = s2 * (1.f / 256.f) - mu * mu;
  const float rs  = rsqrtf(var + 1e-5f);
  __bf16* orow = xn + ((size_t)(b * LSEQ + l0 + r)) * DMODEL;
#pragma unroll
  for (int j = 0; j < 32; ++j) {
    const int c = cb + j;
    orow[c] = (__bf16)((sh[r * 257 + c] - mu) * rs * nw[c] + nb[c]);
  }
}

// Causal depthwise conv (k=4) + bias + SiLU over xin = xz[:, :512].
__global__ void conv_silu(const float* __restrict__ xz,
                          const float* __restrict__ cw,
                          const float* __restrict__ cb,
                          float* __restrict__ xcf, __bf16* __restrict__ xcb) {
  const int idx = blockIdx.x * blockDim.x + threadIdx.x;
  if (idx >= MROWS * DINNER) return;
  const int d = idx & (DINNER - 1);
  const int m = idx >> 9;
  const int l = m & (LSEQ - 1);
  float acc = cb[d];
#pragma unroll
  for (int j = 0; j < 4; ++j) {
    const int ll = l - 3 + j;
    if (ll >= 0)
      acc += cw[d * 4 + j] * xz[(size_t)(m - 3 + j) * (2 * DINNER) + d];
  }
  const float v = acc / (1.f + __expf(-acc));
  xcf[idx] = v;
  xcb[idx] = (__bf16)v;
}

// Pack dbl[:, :16] into K=32 zero-padded bf16 for the dt GEMM.
__global__ void pack_dtin(const float* __restrict__ dbl, __bf16* __restrict__ dtin) {
  const int idx = blockIdx.x * blockDim.x + threadIdx.x;
  if (idx >= MROWS * 32) return;
  const int m = idx >> 5, r = idx & 31;
  dtin[idx] = (__bf16)(r < 16 ? dbl[(size_t)m * 48 + r] : 0.f);
}

__global__ void softplus_bias(float* __restrict__ dt, const float* __restrict__ bias) {
  const int idx = blockIdx.x * blockDim.x + threadIdx.x;
  if (idx >= MROWS * DINNER) return;
  const float t = dt[idx] + bias[idx & (DINNER - 1)];
  dt[idx] = (t > 20.f) ? t : log1pf(__expf(t));
}

// Precompute -exp(A_log) * log2(e) so the scan uses native v_exp (base-2).
__global__ void prep_A(const float* __restrict__ A_log, float* __restrict__ An) {
  const int idx = blockIdx.x * blockDim.x + threadIdx.x;
  if (idx >= DINNER * DSTATE) return;
  An[idx] = -__expf(A_log[idx]) * 1.4426950408889634f;
}

// Selective scan: 16 blocks (4 batches x 4 channel-chunks) x 128 lanes.
// Each lane owns one (b,d) channel's 16-state recurrence in registers;
// B/C (shared across the batch's channels) staged in LDS each step.
__global__ void selective_scan(const float* __restrict__ dt,
                               const float* __restrict__ u,
                               const float* __restrict__ dbl,
                               const float* __restrict__ An,
                               const float* __restrict__ Dp,
                               float* __restrict__ y) {
  __shared__ float sBC[32];
  const int b = blockIdx.x >> 2;
  const int d = ((blockIdx.x & 3) << 7) + threadIdx.x;
  float a2[DSTATE], h[DSTATE];
#pragma unroll
  for (int n = 0; n < DSTATE; ++n) { a2[n] = An[d * DSTATE + n]; h[n] = 0.f; }
  const float Dd = Dp[d];
  size_t m = (size_t)b * LSEQ;
  for (int l = 0; l < LSEQ; ++l, ++m) {
    __syncthreads();
    if (threadIdx.x < 32) sBC[threadIdx.x] = dbl[m * 48 + 16 + threadIdx.x];
    __syncthreads();
    if (l + 1 < LSEQ) __builtin_prefetch(&dbl[(m + 1) * 48 + 16], 0, 1);
    const float dtv = dt[m * DINNER + d];
    const float uv  = u[m * DINNER + d];
    const float du  = dtv * uv;
    float yv = 0.f;
#pragma unroll
    for (int n = 0; n < DSTATE; ++n) {
      h[n] = h[n] * exp2f(dtv * a2[n]) + du * sBC[n];
      yv  += h[n] * sBC[16 + n];
    }
    y[m * DINNER + d] = yv + uv * Dd;
  }
}

// y *= silu(z);  z = xz[:, 512:1024].  Emit bf16 for out_proj GEMM.
__global__ void gate_silu(const float* __restrict__ yb, const float* __restrict__ xz,
                          __bf16* __restrict__ ybf) {
  const int idx = blockIdx.x * blockDim.x + threadIdx.x;
  if (idx >= MROWS * DINNER) return;
  const int d = idx & (DINNER - 1);
  const int m = idx >> 9;
  const float z = xz[(size_t)m * (2 * DINNER) + DINNER + d];
  ybf[idx] = (__bf16)(yb[idx] * z / (1.f + __expf(-z)));
}

__global__ void cvt_bf16(const float* __restrict__ s, __bf16* __restrict__ d, int n) {
  const int idx = blockIdx.x * blockDim.x + threadIdx.x;
  if (idx < n) d[idx] = (__bf16)s[idx];
}

// dt_proj_w (512x16) -> zero-padded (512x32) bf16.
__global__ void cvt_dtw(const float* __restrict__ s, __bf16* __restrict__ d) {
  const int idx = blockIdx.x * blockDim.x + threadIdx.x;
  if (idx >= DINNER * 32) return;
  const int r = idx & 31, q = idx >> 5;
  d[idx] = (__bf16)(r < 16 ? s[q * 16 + r] : 0.f);
}

extern "C" void kernel_launch(void* const* d_in, const int* in_sizes, int n_in,
                              void* d_out, int out_size, void* d_ws, size_t ws_size,
                              hipStream_t stream) {
  (void)in_sizes; (void)n_in; (void)out_size;
  const float* x      = (const float*)d_in[0];
  const float* norm_w = (const float*)d_in[1];
  const float* norm_b = (const float*)d_in[2];
  const float* w_in   = (const float*)d_in[3];
  const float* conv_w = (const float*)d_in[4];
  const float* conv_b = (const float*)d_in[5];
  const float* w_xp   = (const float*)d_in[6];
  const float* w_dt   = (const float*)d_in[7];
  const float* dt_b   = (const float*)d_in[8];
  const float* A_log  = (const float*)d_in[9];
  const float* D_ssm  = (const float*)d_in[10];
  const float* w_out  = (const float*)d_in[11];
  const float* w_pr   = (const float*)d_in[12];
  const float* pr_b   = (const float*)d_in[13];
  const float* skip   = (const float*)d_in[14];
  float* out = (float*)d_out;

  char* ws = (char*)d_ws;
  size_t off = 0;
  auto alloc = [&](size_t bytes) -> void* {
    void* p = ws + off;
    off += (bytes + 255) & ~(size_t)255;
    return p;
  };
  __bf16* w_in_b  = (__bf16*)alloc((size_t)2 * DINNER * DMODEL * 2);    // (1024,256)
  __bf16* w_xp_b  = (__bf16*)alloc((size_t)48 * DINNER * 2);            // (48,512)
  __bf16* w_dt_b  = (__bf16*)alloc((size_t)DINNER * 32 * 2);            // (512,32) padded
  __bf16* w_out_b = (__bf16*)alloc((size_t)DMODEL * DINNER * 2);        // (256,512)
  __bf16* w_pr_b  = (__bf16*)alloc((size_t)DMODEL * DMODEL * 2);        // (256,256)
  float*  Aneg    = (float*)alloc((size_t)DINNER * DSTATE * 4);
  __bf16* xn_b    = (__bf16*)alloc((size_t)MROWS * DMODEL * 2);
  float*  xz      = (float*)alloc((size_t)MROWS * 2 * DINNER * 4);
  float*  xc_f    = (float*)alloc((size_t)MROWS * DINNER * 4);
  __bf16* xc_b    = (__bf16*)alloc((size_t)MROWS * DINNER * 2);
  float*  dbl     = (float*)alloc((size_t)MROWS * 48 * 4);
  __bf16* dtin_b  = (__bf16*)alloc((size_t)MROWS * 32 * 2);
  float*  dtbuf   = (float*)alloc((size_t)MROWS * DINNER * 4);
  float*  ybuf    = (float*)alloc((size_t)MROWS * DINNER * 4);
  __bf16* y_b     = (__bf16*)alloc((size_t)MROWS * DINNER * 2);
  __bf16* xm_b    = (__bf16*)alloc((size_t)MROWS * DMODEL * 2);
  if (off > ws_size) return;  // workspace too small; nothing safe to do

  const int T = 256;
  // Weight conversion + A precompute
  cvt_bf16<<<(2*DINNER*DMODEL + T-1)/T, T, 0, stream>>>(w_in,  w_in_b,  2*DINNER*DMODEL);
  cvt_bf16<<<(48*DINNER       + T-1)/T, T, 0, stream>>>(w_xp,  w_xp_b,  48*DINNER);
  cvt_dtw <<<(DINNER*32       + T-1)/T, T, 0, stream>>>(w_dt,  w_dt_b);
  cvt_bf16<<<(DMODEL*DINNER   + T-1)/T, T, 0, stream>>>(w_out, w_out_b, DMODEL*DINNER);
  cvt_bf16<<<(DMODEL*DMODEL   + T-1)/T, T, 0, stream>>>(w_pr,  w_pr_b,  DMODEL*DMODEL);
  prep_A  <<<(DINNER*DSTATE   + T-1)/T, T, 0, stream>>>(A_log, Aneg);

  // LayerNorm (NCHW transpose in LDS) -> bf16 tokens
  layernorm_to_bf16<<<dim3(LSEQ/32, BSZ), 256, 0, stream>>>(x, norm_w, norm_b, xn_b);

  // in_proj: (16384,256) x (1024,256)^T -> xz (16384,1024)
  gemm_bf16_f32<<<dim3(2*DINNER/16, MROWS/64), 128, 0, stream>>>(
      xn_b, DMODEL, w_in_b, DMODEL, xz, 2*DINNER, DMODEL);

  // causal conv + SiLU
  conv_silu<<<(MROWS*DINNER + T-1)/T, T, 0, stream>>>(xz, conv_w, conv_b, xc_f, xc_b);

  // x_proj: (16384,512) x (48,512)^T -> dbl (16384,48)
  gemm_bf16_f32<<<dim3(48/16, MROWS/64), 128, 0, stream>>>(
      xc_b, DINNER, w_xp_b, DINNER, dbl, 48, DINNER);

  // dt_proj: pad K to 32, GEMM, softplus(+bias)
  pack_dtin<<<(MROWS*32 + T-1)/T, T, 0, stream>>>(dbl, dtin_b);
  gemm_bf16_f32<<<dim3(DINNER/16, MROWS/64), 128, 0, stream>>>(
      dtin_b, 32, w_dt_b, 32, dtbuf, DINNER, 32);
  softplus_bias<<<(MROWS*DINNER + T-1)/T, T, 0, stream>>>(dtbuf, dt_b);

  // selective scan: all 2048 channels in flight
  selective_scan<<<BSZ*4, 128, 0, stream>>>(dtbuf, xc_f, dbl, Aneg, D_ssm, ybuf);

  // gate with silu(z), emit bf16
  gate_silu<<<(MROWS*DINNER + T-1)/T, T, 0, stream>>>(ybuf, xz, y_b);

  // out_proj + skip (fused), proj + bias + transposed store (fused)
  gemm_outproj<<<dim3(DMODEL/16, MROWS/64), 128, 0, stream>>>(y_b, w_out_b, x, skip, xm_b);
  gemm_proj   <<<dim3(DMODEL/16, MROWS/64), 128, 0, stream>>>(xm_b, w_pr_b, pr_b, out);
}
